// GP_8701603741721
// MI455X (gfx1250) — matrix operand order repaired and checked by reading, not compiled, not run on hardware
//
#include <hip/hip_runtime.h>

// ---------------------------------------------------------------------------
// Types for CDNA5 WMMA
// ---------------------------------------------------------------------------
typedef __attribute__((ext_vector_type(16))) __bf16        v16bf;
typedef __attribute__((ext_vector_type(8)))  float         v8f;
typedef __attribute__((ext_vector_type(4)))  unsigned int  uivec4;
typedef __attribute__((ext_vector_type(2)))  unsigned int  uivec2;

#define BM 128
#define BN 128
#define BK 32
#define NTHR 256

union FragBF { v16bf v; uivec4 q[2]; };

__device__ __forceinline__ unsigned short bf16_rn(float f) {
  unsigned u = __float_as_uint(f);
  u += 0x7fffu + ((u >> 16) & 1u);
  return (unsigned short)(u >> 16);
}

// ---------------------------------------------------------------------------
// f32 -> (bf16 hi, bf16 lo) split with arbitrary source strides.
// dst is row-major R x C;  dst(r,c) = src[r*srs + c*scs]
// ---------------------------------------------------------------------------
__global__ void __launch_bounds__(256) k_conv_split(
    const float* __restrict__ src, unsigned short* __restrict__ hi,
    unsigned short* __restrict__ lo, int R, int C, long long srs, long long scs)
{
  long long idx = (long long)blockIdx.x * 256 + threadIdx.x;
  long long total = (long long)R * C;
  if (idx >= total) return;
  int r = (int)(idx / C), c = (int)(idx % C);
  float x = src[(long long)r * srs + (long long)c * scs];
  unsigned u = __float_as_uint(x);
  unsigned short h = (unsigned short)(u >> 16);     // truncated hi part
  float hf = __uint_as_float(u & 0xffff0000u);
  hi[idx] = h;
  lo[idx] = bf16_rn(x - hf);                        // residual, RN
}

// ---------------------------------------------------------------------------
// Split-bf16 WMMA GEMM:  D(MxN) = alpha * A(MxK) @ B(KxN) + beta * Csrc
// A given as hi/lo bf16 row-major MxK ; B as hi/lo bf16 row-major KxN.
// Requires M % 128 == 0 and K % 32 == 0 (true for all uses here); N is
// guarded (N = 64 for the thin GEMMs).
// A tiles are staged with GLOBAL_LOAD_ASYNC_TO_LDS_B128 (ASYNCcnt path);
// B tiles are transposed into LDS so fragments are contiguous b128 reads.
// ---------------------------------------------------------------------------
__global__ void __launch_bounds__(NTHR) k_gemm_split(
    const unsigned short* __restrict__ Ahi, const unsigned short* __restrict__ Alo,
    const unsigned short* __restrict__ Bhi, const unsigned short* __restrict__ Blo,
    const float* __restrict__ Csrc, float* __restrict__ D,
    int M, int N, int K, float alpha, float beta)
{
  __shared__ __align__(16) unsigned short lAh[BM * BK];
  __shared__ __align__(16) unsigned short lAl[BM * BK];
  __shared__ __align__(16) unsigned short lBh[BN * BK];   // stored transposed [n][k]
  __shared__ __align__(16) unsigned short lBl[BN * BK];

  const int tid  = threadIdx.x;
  const int lane = tid & 31;
  const int w    = tid >> 5;
  const int wm   = w >> 1;           // 0..3 -> 32-row band
  const int wn   = w & 1;            // 0..1 -> 64-col band
  const long long m0 = (long long)blockIdx.y * BM;
  const long long n0 = (long long)blockIdx.x * BN;

  // LDS byte offsets of the staging buffers (flat addr low 32 bits == LDS offset)
  const unsigned aBaseH = (unsigned)(size_t)(void*)lAh;
  const unsigned aBaseL = (unsigned)(size_t)(void*)lAl;

  v8f acc[2][4];
#pragma unroll
  for (int mt = 0; mt < 2; ++mt)
#pragma unroll
    for (int nt = 0; nt < 4; ++nt)
#pragma unroll
      for (int r = 0; r < 8; ++r) acc[mt][nt][r] = 0.0f;

  for (int k0 = 0; k0 < K; k0 += BK) {
    __syncthreads();
    // ---- stage A tile (BM x BK) hi/lo via async global->LDS DMA (b128) ----
#pragma unroll
    for (int i = 0; i < 2; ++i) {
      int idx = i * NTHR + tid;          // 0..511 chunks of 8 halfs
      int r   = idx >> 2;                // 0..127
      int c8  = (idx & 3) << 3;          // 0,8,16,24
      long long g = (m0 + r) * (long long)K + k0 + c8;
      unsigned lof = (unsigned)((r * BK + c8) * 2);
      unsigned long long gah = (unsigned long long)(const void*)(Ahi + g);
      unsigned long long gal = (unsigned long long)(const void*)(Alo + g);
      asm volatile("global_load_async_to_lds_b128 %0, %1, off"
                   :: "v"(aBaseH + lof), "v"(gah) : "memory");
      asm volatile("global_load_async_to_lds_b128 %0, %1, off"
                   :: "v"(aBaseL + lof), "v"(gal) : "memory");
    }
    // ---- stage B tile (BK x BN) transposed into LDS as [n][k] ----
#pragma unroll
    for (int i = 0; i < 4; ++i) {
      int idx = i * NTHR + tid;
      int kk  = idx >> 5;                // 0..31
      int nc  = (idx & 31) << 2;         // 0..124 (4-half groups)
      long long gB = (long long)(k0 + kk) * N + n0 + nc;
      union { uivec2 v; unsigned short s[4]; } th, tl;
      if ((int)n0 + nc + 3 < N) {        // all-or-nothing per 4-half group
        th.v = *(const uivec2*)(Bhi + gB);
        tl.v = *(const uivec2*)(Blo + gB);
        if (k0 + BK < K) {
          __builtin_prefetch(Bhi + gB + (long long)BK * N, 0, 1);  // global_prefetch_b8
          __builtin_prefetch(Blo + gB + (long long)BK * N, 0, 1);
        }
      } else {
#pragma unroll
        for (int q = 0; q < 4; ++q) { th.s[q] = 0; tl.s[q] = 0; }
      }
#pragma unroll
      for (int q = 0; q < 4; ++q) {
        lBh[(nc + q) * BK + kk] = th.s[q];
        lBl[(nc + q) * BK + kk] = tl.s[q];
      }
    }
    asm volatile("s_wait_asynccnt 0x0" ::: "memory");   // A tiles landed in LDS
    __syncthreads();

    // ---- build fragments per documented CDNA5 16-bit A/B VGPR layouts ----
    const int h  = lane >> 4;
    const int ml = lane & 15;
    FragBF ah[2], al[2], bh[4], bl[4];
#pragma unroll
    for (int mt = 0; mt < 2; ++mt) {
      int m = wm * 32 + mt * 16 + ml;
      ah[mt].q[0] = *(const uivec4*)&lAh[m * BK + 8 * h];        // K = 8h..8h+7
      ah[mt].q[1] = *(const uivec4*)&lAh[m * BK + 16 + 8 * h];   // K = 16+8h..
      al[mt].q[0] = *(const uivec4*)&lAl[m * BK + 8 * h];
      al[mt].q[1] = *(const uivec4*)&lAl[m * BK + 16 + 8 * h];
    }
#pragma unroll
    for (int nt = 0; nt < 4; ++nt) {
      int nn = wn * 64 + nt * 16 + ml;
      bh[nt].q[0] = *(const uivec4*)&lBh[nn * BK + 8 * h];       // K = 8h..8h+7
      bh[nt].q[1] = *(const uivec4*)&lBh[nn * BK + 16 + 8 * h];  // K = 16+8h..
      bl[nt].q[0] = *(const uivec4*)&lBl[nn * BK + 8 * h];
      bl[nt].q[1] = *(const uivec4*)&lBl[nn * BK + 16 + 8 * h];
    }

    // ---- split-bf16 product: hi*hi + hi*lo + lo*hi (~f32 accuracy) ----
#pragma unroll
    for (int mt = 0; mt < 2; ++mt)
#pragma unroll
      for (int nt = 0; nt < 4; ++nt) {
        v8f c = acc[mt][nt];
        c = __builtin_amdgcn_wmma_f32_16x16x32_bf16(false, ah[mt].v, false, bh[nt].v, (short)0, c, false, false);
        c = __builtin_amdgcn_wmma_f32_16x16x32_bf16(false, ah[mt].v, false, bl[nt].v, (short)0, c, false, false);
        c = __builtin_amdgcn_wmma_f32_16x16x32_bf16(false, al[mt].v, false, bh[nt].v, (short)0, c, false, false);
        acc[mt][nt] = c;
      }
  }

  // ---- epilogue: C/D layout -> m = base + r + 8*(lane>=16), n = base + lane%16
#pragma unroll
  for (int mt = 0; mt < 2; ++mt)
#pragma unroll
    for (int nt = 0; nt < 4; ++nt) {
      long long mb = m0 + wm * 32 + mt * 16;
      int nb = (int)n0 + wn * 64 + nt * 16 + (lane & 15);
#pragma unroll
      for (int r = 0; r < 8; ++r) {
        long long m = mb + r + 8 * (lane >> 4);
        if (nb < N) {
          long long o = m * (long long)N + nb;
          float v = alpha * acc[mt][nt][r];
          if (beta != 0.0f) v += beta * Csrc[o];
          D[o] = v;
        }
      }
    }
}

// ---------------------------------------------------------------------------
// Column norms of (C x n) matrix: nrm[m] = ||x[:, m]||  (coalesced across m)
// ---------------------------------------------------------------------------
__global__ void __launch_bounds__(256) k_colnorm(
    const float* __restrict__ x, float* __restrict__ nrm, int c, int n)
{
  int m = blockIdx.x * 256 + threadIdx.x;
  if (m >= n) return;
  float s = 0.f;
  for (int k = 0; k < c; ++k) { float v = x[(long long)k * n + m]; s += v * v; }
  nrm[m] = sqrtf(s);
}

// S -> exp(S/(na*nb+eps) - 1) (+ diag) in place
__global__ void __launch_bounds__(256) k_kernelize(
    float* __restrict__ S, const float* __restrict__ na,
    const float* __restrict__ nb, int n, float diag)
{
  long long idx = (long long)blockIdx.x * 256 + threadIdx.x;
  if (idx >= (long long)n * n) return;
  int m = (int)(idx / n), q = (int)(idx % n);
  float v = S[idx] / (na[m] * nb[q] + 1e-6f);
  v = __expf(v - 1.0f);                 // T_KER = 1.0
  if (m == q) v += diag;                // +SIGMA on K_yy
  S[idx] = v;
}

__global__ void __launch_bounds__(256) k_rowsum(
    const float* __restrict__ A, float* __restrict__ rs, int n)
{
  __shared__ float sm[256];
  int row = blockIdx.x;
  float s = 0.f;
  for (int k = threadIdx.x; k < n; k += 256) s += fabsf(A[(long long)row * n + k]);
  sm[threadIdx.x] = s; __syncthreads();
  for (int st = 128; st > 0; st >>= 1) {
    if (threadIdx.x < st) sm[threadIdx.x] += sm[threadIdx.x + st];
    __syncthreads();
  }
  if (threadIdx.x == 0) rs[row] = sm[0];
}

__global__ void __launch_bounds__(256) k_maxscale(
    const float* __restrict__ rs, float* __restrict__ sc, int n)
{
  __shared__ float sm[256];
  float m = 0.f;
  for (int i = threadIdx.x; i < n; i += 256) m = fmaxf(m, rs[i]);
  sm[threadIdx.x] = m; __syncthreads();
  for (int st = 128; st > 0; st >>= 1) {
    if (threadIdx.x < st) sm[threadIdx.x] = fmaxf(sm[threadIdx.x], sm[threadIdx.x + st]);
    __syncthreads();
  }
  if (threadIdx.x == 0) sc[0] = 1.0f / (sm[0] * sm[0] + 1e-30f);  // X0 = A / (||A||1 * ||A||inf)
}

__global__ void __launch_bounds__(256) k_scale(
    float* __restrict__ dst, const float* __restrict__ src,
    const float* __restrict__ sc, long long total)
{
  long long idx = (long long)blockIdx.x * 256 + threadIdx.x;
  if (idx >= total) return;
  dst[idx] = src[idx] * sc[0];
}

// positional features ff[p*d + dd] = cos(8*pi*(w0*gx + w1*gy + b))
__global__ void __launch_bounds__(256) k_posfeat(
    const float* __restrict__ pw, const float* __restrict__ pb,
    float* __restrict__ ff, int h, int w, int d)
{
  int idx = blockIdx.x * 256 + threadIdx.x;
  if (idx >= h * w * d) return;
  int p = idx / d, dd = idx % d;
  int i = p / w, j = p % w;
  float gy = (2.f * i + 1.f) / (float)h - 1.f;
  float gx = (2.f * j + 1.f) / (float)w - 1.f;
  float ph = pw[dd * 2 + 0] * gx + pw[dd * 2 + 1] * gy + pb[dd];
  ff[idx] = cosf(8.0f * 3.14159265358979323846f * ph);
}

// mu (n x d) -> out channels 0..d-1 (channel-major)
__global__ void __launch_bounds__(256) k_mu_out(
    const float* __restrict__ mu, float* __restrict__ outb, int n, int d)
{
  int idx = blockIdx.x * 256 + threadIdx.x;
  if (idx >= n * d) return;
  int p = idx / d, dd = idx % d;
  outb[(long long)dd * n + p] = mu[idx];
}

// 5x5 local covariance gather: cov is (n x n) with n = h*w
__global__ void __launch_bounds__(256) k_local_out(
    const float* __restrict__ cov, float* __restrict__ outb, int h, int w)
{
  int n = h * w;
  int idx = blockIdx.x * 256 + threadIdx.x;
  if (idx >= n * 25) return;
  int p = idx / 25, kk = idx % 25;
  int r = p / w, c = p % w;
  int dr = kk / 5 - 2, dc = kk % 5 - 2;
  int rr = r + dr, cc = c + dc;
  float v = 0.f;
  if (rr >= 0 && rr < h && cc >= 0 && cc < w)
    v = cov[(long long)p * n + (long long)rr * w + cc];
  outb[(long long)kk * n + p] = v;
}

// ---------------------------------------------------------------------------
// Host orchestration
// ---------------------------------------------------------------------------
extern "C" void kernel_launch(void* const* d_in, const int* in_sizes, int n_in,
                              void* d_out, int out_size, void* d_ws, size_t ws_size,
                              hipStream_t stream)
{
  (void)in_sizes; (void)n_in; (void)out_size; (void)ws_size;
  const float* xin = (const float*)d_in[0];   // (2, 512, 64, 64)
  const float* yin = (const float*)d_in[1];   // (2, 512, 64, 64)
  const float* pw  = (const float*)d_in[2];   // (64, 2)
  const float* pb  = (const float*)d_in[3];   // (64,)
  float* out = (float*)d_out;                 // (2, 89, 64, 64)

  const int B = 2, C = 512, H = 64, W = 64, Dd = 64;
  const int n = H * W;                        // 4096
  const long long N2 = (long long)n * n;
  const int NITER = 20;                       // Newton-Schulz iterations

  char* ws = (char*)d_ws;
  size_t off = 0;
  auto alloc = [&](size_t bytes) -> void* {
    void* p = (void*)(ws + off);
    off += (bytes + 255) & ~(size_t)255;
    return p;
  };

  // big f32 buffers (64 MB each)
  float* KXY = (float*)alloc(N2 * 4);
  float* KYY = (float*)alloc(N2 * 4);
  float* KXX = (float*)alloc(N2 * 4);
  float* Xb  = (float*)alloc(N2 * 4);
  float* X2b = (float*)alloc(N2 * 4);
  float* Wb  = (float*)alloc(N2 * 4);
  // big bf16 hi/lo buffers (32 MB each)
  unsigned short* AH = (unsigned short*)alloc(N2 * 2);
  unsigned short* AL = (unsigned short*)alloc(N2 * 2);
  unsigned short* PH = (unsigned short*)alloc(N2 * 2);
  unsigned short* PL = (unsigned short*)alloc(N2 * 2);
  unsigned short* QH = (unsigned short*)alloc(N2 * 2);
  unsigned short* QL = (unsigned short*)alloc(N2 * 2);
  // input conversions (n x C and C x n)
  size_t nc2 = (size_t)n * C * 2;
  unsigned short* XAH = (unsigned short*)alloc(nc2);
  unsigned short* XAL = (unsigned short*)alloc(nc2);
  unsigned short* XBH = (unsigned short*)alloc(nc2);
  unsigned short* XBL = (unsigned short*)alloc(nc2);
  unsigned short* YAH = (unsigned short*)alloc(nc2);
  unsigned short* YAL = (unsigned short*)alloc(nc2);
  unsigned short* YBH = (unsigned short*)alloc(nc2);
  unsigned short* YBL = (unsigned short*)alloc(nc2);
  // small
  float* NX = (float*)alloc(n * 4);
  float* NY = (float*)alloc(n * 4);
  float* RS = (float*)alloc(n * 4);
  float* SC = (float*)alloc(256);
  float* FF = (float*)alloc((size_t)n * Dd * 4);
  unsigned short* FFH = (unsigned short*)alloc((size_t)n * Dd * 2);
  unsigned short* FFL = (unsigned short*)alloc((size_t)n * Dd * 2);
  float* Zf = (float*)alloc((size_t)n * Dd * 4);
  unsigned short* ZH = (unsigned short*)alloc((size_t)n * Dd * 2);
  unsigned short* ZL = (unsigned short*)alloc((size_t)n * Dd * 2);
  float* MU = (float*)alloc((size_t)n * Dd * 4);

  dim3 gNN2((n + BN - 1) / BN, (n + BM - 1) / BM);   // 32 x 32 blocks
  dim3 gND((Dd + BN - 1) / BN, (n + BM - 1) / BM);   // 1 x 32 blocks
  int eNN  = (int)((N2 + 255) / 256);
  int eNC  = (int)(((long long)n * C + 255) / 256);
  int eND  = (n * Dd + 255) / 256;

  // positional features (batch independent)
  k_posfeat<<<eND, 256, 0, stream>>>(pw, pb, FF, H, W, Dd);
  k_conv_split<<<eND, 256, 0, stream>>>(FF, FFH, FFL, n, Dd, Dd, 1);

  for (int b = 0; b < B; ++b) {
    const float* xb = xin + (long long)b * C * n;
    const float* yb = yin + (long long)b * C * n;
    float* outb = out + (long long)b * 89 * n;

    k_colnorm<<<(n + 255) / 256, 256, 0, stream>>>(xb, NX, C, n);
    k_colnorm<<<(n + 255) / 256, 256, 0, stream>>>(yb, NY, C, n);

    // xf = x^T (n x C) as A-operand; x (C x n) as B-operand; same for y
    k_conv_split<<<eNC, 256, 0, stream>>>(xb, XAH, XAL, n, C, 1, n);
    k_conv_split<<<eNC, 256, 0, stream>>>(xb, XBH, XBL, C, n, n, 1);
    k_conv_split<<<eNC, 256, 0, stream>>>(yb, YAH, YAL, n, C, 1, n);
    k_conv_split<<<eNC, 256, 0, stream>>>(yb, YBH, YBL, C, n, n, 1);

    // Gram matrices (K = 512)
    k_gemm_split<<<gNN2, NTHR, 0, stream>>>(XAH, XAL, YBH, YBL, KXY, KXY, n, n, C, 1.f, 0.f);
    k_gemm_split<<<gNN2, NTHR, 0, stream>>>(YAH, YAL, YBH, YBL, KYY, KYY, n, n, C, 1.f, 0.f);
    k_gemm_split<<<gNN2, NTHR, 0, stream>>>(XAH, XAL, XBH, XBL, KXX, KXX, n, n, C, 1.f, 0.f);

    // cosine kernelization (KYY gets +SIGMA on diagonal)
    k_kernelize<<<eNN, 256, 0, stream>>>(KXY, NX, NY, n, 0.0f);
    k_kernelize<<<eNN, 256, 0, stream>>>(KYY, NY, NY, n, 0.1f);
    k_kernelize<<<eNN, 256, 0, stream>>>(KXX, NX, NX, n, 0.0f);

    // A = split(KYY_plus); X0 = A / (maxrowsum^2)
    k_conv_split<<<eNN, 256, 0, stream>>>(KYY, AH, AL, n, n, n, 1);
    k_rowsum<<<n, 256, 0, stream>>>(KYY, RS, n);
    k_maxscale<<<1, 256, 0, stream>>>(RS, SC, n);
    float* X  = Xb;
    float* Xn = X2b;
    k_scale<<<eNN, 256, 0, stream>>>(X, KYY, SC, N2);

    // Newton-Schulz: X <- 2X - X (A X)
    for (int it = 0; it < NITER; ++it) {
      k_conv_split<<<eNN, 256, 0, stream>>>(X, PH, PL, n, n, n, 1);
      k_gemm_split<<<gNN2, NTHR, 0, stream>>>(AH, AL, PH, PL, Wb, Wb, n, n, n, 1.f, 0.f);
      k_conv_split<<<eNN, 256, 0, stream>>>(Wb, QH, QL, n, n, n, 1);
      k_gemm_split<<<gNN2, NTHR, 0, stream>>>(PH, PL, QH, QL, X, Xn, n, n, n, -1.f, 2.f);
      float* t = X; X = Xn; Xn = t;
    }

    // P = split(X_final), Q = split(K_xy)
    k_conv_split<<<eNN, 256, 0, stream>>>(X, PH, PL, n, n, n, 1);
    k_conv_split<<<eNN, 256, 0, stream>>>(KXY, QH, QL, n, n, n, 1);

    // Z = Kyy_inv @ ff ; mu = K_xy @ Z
    k_gemm_split<<<gND, NTHR, 0, stream>>>(PH, PL, FFH, FFL, Zf, Zf, n, Dd, n, 1.f, 0.f);
    k_conv_split<<<eND, 256, 0, stream>>>(Zf, ZH, ZL, n, Dd, Dd, 1);
    k_gemm_split<<<gND, NTHR, 0, stream>>>(QH, QL, ZH, ZL, MU, MU, n, Dd, n, 1.f, 0.f);
    k_mu_out<<<eND, 256, 0, stream>>>(MU, outb, n, Dd);

    // U = K_xy @ Kyy_inv ; cov = K_xx - U @ K_xy^T
    k_gemm_split<<<gNN2, NTHR, 0, stream>>>(QH, QL, PH, PL, Wb, Wb, n, n, n, 1.f, 0.f);
    k_conv_split<<<eNN, 256, 0, stream>>>(Wb, PH, PL, n, n, n, 1);       // P = split(U)
    k_conv_split<<<eNN, 256, 0, stream>>>(KXY, AH, AL, n, n, 1, n);      // A = split(K_xy^T)
    k_gemm_split<<<gNN2, NTHR, 0, stream>>>(PH, PL, AH, AL, KXX, Xn, n, n, n, -1.f, 1.f);

    // 5x5 local covariance -> out channels 64..88
    k_local_out<<<(n * 25 + 255) / 256, 256, 0, stream>>>(Xn, outb + (long long)64 * n, H, W);
  }
}
